// NeuralMemoryModule_20744692040327
// MI455X (gfx1250) — matrix operand order repaired and checked by reading, compiled, not run on hardware
//
#include <hip/hip_runtime.h>
#include <hip/hip_bf16.h>

#define BATCH 64
#define SEQ   128
#define DIM   512

// LDS B-tile chunk: 64 rows x 32 k-elems, row stride padded to 40 elems (80B)
// -> per-lane ds_load stride = 20 dwords, gcd(20,64)=4 -> conflict-free b128 reads.
#define CH_ROW   40
#define CH_ELEMS (64 * CH_ROW)   // 2560 u16 per buffer

typedef unsigned short u16t;
typedef __attribute__((ext_vector_type(16))) __bf16 v16bf;
typedef __attribute__((ext_vector_type(16))) unsigned short v16u;
typedef __attribute__((ext_vector_type(8)))  float v8f;

// ---------- scalar helpers ----------
__device__ __forceinline__ u16t f2bf(float f) {
  unsigned int u = __float_as_uint(f);
  u += 0x7fffu + ((u >> 16) & 1u);          // round-to-nearest-even
  return (u16t)(u >> 16);
}
__device__ __forceinline__ float bf2f(u16t h) {
  return __uint_as_float(((unsigned int)h) << 16);
}
__device__ __forceinline__ float sigmoidf(float z) {
  return 1.0f / (1.0f + __expf(-z));
}
__device__ __forceinline__ void zero8(v8f& a) {
#pragma unroll
  for (int e = 0; e < 8; ++e) a[e] = 0.0f;
}

// ---------- WMMA fragment loaders (wave32, 16x16x32 bf16) ----------
// A (16x32): lane L holds row m = L%16; half=L/16.
//   element e -> k = (e<8 ? e : e+8) + 8*half   (two contiguous 8-elem runs)
__device__ __forceinline__ v16bf ld_afrag(const u16t* row, int k0, int half) {
  v16u r;
  const u16t* p0 = row + k0 + 8 * half;
#pragma unroll
  for (int e = 0; e < 8; ++e) { r[e] = p0[e]; r[e + 8] = p0[e + 16]; }
  return __builtin_bit_cast(v16bf, r);
}
// B (32x16) from an LDS-staged chunk row (contiguous along k within the chunk):
// lane L holds col n = L%16; element e -> k = (e<8 ? e : e+8) + 8*half
__device__ __forceinline__ v16bf ld_bfrag_row(const u16t* row, int half) {
  v16u r;
  const u16t* p = row + 8 * half;
#pragma unroll
  for (int e = 0; e < 8; ++e) { r[e] = p[e]; r[e + 8] = p[e + 16]; }
  return __builtin_bit_cast(v16bf, r);
}

// ---------- async global->LDS staging of one B k-chunk ----------
// Chunk c covers k in [c*32, c*32+32) for the block's 64 B-rows (cols n0..n0+63).
// 256 x 16B segments; each thread issues 256/NT async-DMA instructions.
template <int NT>
__device__ __forceinline__ void stage_chunk(const u16t* Bm, int ldb, int n0, int c,
                                            u16t* buf, int tid) {
#pragma unroll
  for (int s = tid; s < 256; s += NT) {
    int row = s >> 2, part = s & 3;
    const u16t* g = Bm + (size_t)(n0 + row) * (size_t)ldb + c * 32 + part * 8;
    unsigned lds = (unsigned)(size_t)(buf + row * CH_ROW + part * 8);
    asm volatile("global_load_async_to_lds_b128 %0, %1, off"
                 :: "v"(lds), "v"(g) : "memory");
  }
}

// One wave computes a 16x64 strip of C (4 tiles), contracting over K.
// B is staged block-wide through LDS with an ASYNCcnt double-buffered pipeline;
// A is per-wave from global, software-pipelined one chunk ahead. All four B
// fragments are preloaded so the four WMMAs issue back-to-back (one DS drain
// per chunk instead of four).
template <int NT>
__device__ __forceinline__ void mma_row64_lds(const u16t* arow, const u16t* Bm, int ldb,
                                              int n0, int tid, int K, u16t* smem,
                                              v8f acc[4]) {
  const int lane = tid & 31;
  const int half = lane >> 4;
  const int nl = lane & 15;
  constexpr int IPC = 256 / NT;      // async instructions per wave per chunk
  const int nchunks = K / 32;

  stage_chunk<NT>(Bm, ldb, n0, 0, smem, tid);
  v16bf a = ld_afrag(arow, 0, half);

  for (int c = 0; c < nchunks; ++c) {
    u16t* cur = smem + (c & 1) * CH_ELEMS;
    u16t* nxt = smem + ((c + 1) & 1) * CH_ELEMS;
    bool more = (c + 1 < nchunks);
    if (more) {
      stage_chunk<NT>(Bm, ldb, n0, c + 1, nxt, tid);
      if (c + 2 < nchunks) __builtin_prefetch(arow + (c + 2) * 32, 0, 3);
      asm volatile("s_wait_asynccnt %0" :: "n"(IPC));  // chunk c landed; c+1 in flight
    } else {
      asm volatile("s_wait_asynccnt 0");
    }
    __syncthreads();                                   // publish chunk c block-wide

    // Preload all four B fragments (8 x ds_load_b128, one drain) ...
    v16bf bfr[4];
#pragma unroll
    for (int j = 0; j < 4; ++j)
      bfr[j] = ld_bfrag_row(cur + (j * 16 + nl) * CH_ROW, half);
    // ... overlap next A-chunk global load with the WMMA burst ...
    v16bf an = a;
    if (more) an = ld_afrag(arow, (c + 1) * 32, half);
    // ... then issue the four WMMAs back-to-back.
#pragma unroll
    for (int j = 0; j < 4; ++j) {
      acc[j] = __builtin_amdgcn_wmma_f32_16x16x32_bf16(
          /*neg_a=*/false, a, /*neg_b=*/false, bfr[j],
          /*c_mod=*/(short)0, acc[j], /*reuse_a=*/false, /*reuse_b=*/false);
    }
    __syncthreads();                                   // cur may be re-staged next iter
    a = an;
  }
}

// ======================= precompute kernels =======================

__global__ void k_convert(const float* __restrict__ x,
                          const float* __restrict__ WQ, const float* __restrict__ WK,
                          const float* __restrict__ WV,
                          u16t* Xb, u16t* WQb, u16t* WKb, u16t* WVb) {
  int i = blockIdx.x * 256 + threadIdx.x;
  if (i < BATCH * SEQ * DIM) Xb[i] = f2bf(x[i]);
  if (i < DIM * DIM) { WQb[i] = f2bf(WQ[i]); WKb[i] = f2bf(WK[i]); WVb[i] = f2bf(WV[i]); }
}

__global__ void k_init(const float* __restrict__ W1, const float* __restrict__ W2,
                       const float* __restrict__ b1, const float* __restrict__ b2,
                       float* W1f, float* W2f, float* M1w, float* M2w,
                       u16t* W1b, u16t* W2b, u16t* W2Tb,
                       float* b1f, float* b2f, float* M1b, float* M2b) {
  int i = blockIdx.x * 256 + threadIdx.x;
  if (i < DIM * DIM) {
    float w1 = W1[i], w2 = W2[i];
    W1f[i] = w1; W2f[i] = w2; M1w[i] = 0.0f; M2w[i] = 0.0f;
    W1b[i] = f2bf(w1); W2b[i] = f2bf(w2);
    int r = i / DIM, c = i % DIM;
    W2Tb[(size_t)c * DIM + r] = f2bf(w2);
  }
  if (i < DIM) { b1f[i] = b1[i]; b2f[i] = b2[i]; M1b[i] = 0.0f; M2b[i] = 0.0f; }
}

__global__ void k_gates(const float* __restrict__ x,
                        const float* __restrict__ tw, const float* __restrict__ tb,
                        const float* __restrict__ ew, const float* __restrict__ eb,
                        const float* __restrict__ aw, const float* __restrict__ ab,
                        float* theta, float* eta, float* alpha) {
  int t = blockIdx.x;
  int wv = threadIdx.x >> 5, lane = threadIdx.x & 31;
  __shared__ float sth[8], set_[8], sal[8];
  float a_th = 0.f, a_et = 0.f, a_al = 0.f;
  for (int b = wv; b < BATCH; b += 8) {
    const float* xp = x + ((size_t)b * SEQ + t) * DIM;
    float s0 = 0.f, s1 = 0.f, s2 = 0.f;
    for (int d = lane; d < DIM; d += 32) {
      float v = xp[d];
      s0 += v * tw[d]; s1 += v * ew[d]; s2 += v * aw[d];
    }
#pragma unroll
    for (int off = 16; off; off >>= 1) {
      s0 += __shfl_xor(s0, off); s1 += __shfl_xor(s1, off); s2 += __shfl_xor(s2, off);
    }
    if (lane == 0) {
      a_th += sigmoidf(s0 + tb[0]); a_et += sigmoidf(s1 + eb[0]); a_al += sigmoidf(s2 + ab[0]);
    }
  }
  if (lane == 0) { sth[wv] = a_th; set_[wv] = a_et; sal[wv] = a_al; }
  __syncthreads();
  if (threadIdx.x == 0) {
    float r0 = 0.f, r1 = 0.f, r2 = 0.f;
    for (int i = 0; i < 8; ++i) { r0 += sth[i]; r1 += set_[i]; r2 += sal[i]; }
    theta[t] = r0 / BATCH; eta[t] = r1 / BATCH; alpha[t] = r2 / BATCH;
  }
}

// Q/K/V = X @ W^T for all t. grid (DIM/64, SEQ, 3), 128 threads (4 waves).
__global__ void k_qkv(const u16t* __restrict__ Xb,
                      const u16t* __restrict__ WQb, const u16t* __restrict__ WKb,
                      const u16t* __restrict__ WVb,
                      u16t* Qb, u16t* Kb, u16t* KTb, float* Vf) {
  __shared__ __attribute__((aligned(16))) u16t smem[2 * CH_ELEMS];
  int t = blockIdx.y, sel = blockIdx.z;
  int n0 = blockIdx.x * 64;
  int wv = threadIdx.x >> 5, lane = threadIdx.x & 31;
  int m0 = wv * 16;
  const u16t* Wb = (sel == 0) ? WQb : (sel == 1) ? WKb : WVb;
  int row = m0 + (lane & 15);
  const u16t* arow = Xb + ((size_t)row * SEQ + t) * DIM;
  v8f acc[4];
#pragma unroll
  for (int j = 0; j < 4; ++j) zero8(acc[j]);
  mma_row64_lds<128>(arow, Wb, DIM, n0, threadIdx.x, DIM, smem, acc);
  size_t base = (size_t)t * BATCH * DIM;
  int half = lane >> 4, nl = lane & 15;
#pragma unroll
  for (int j = 0; j < 4; ++j) {
    int n = n0 + j * 16 + nl;
#pragma unroll
    for (int e = 0; e < 8; ++e) {
      int m = m0 + e + 8 * half;
      float v = acc[j][e];
      if (sel == 0) {
        Qb[base + (size_t)m * DIM + n] = f2bf(v);
      } else if (sel == 1) {
        Kb[base + (size_t)m * DIM + n] = f2bf(v);
        KTb[base + (size_t)n * BATCH + m] = f2bf(v);   // k^T for gW1
      } else {
        Vf[base + (size_t)m * DIM + n] = v;
      }
    }
  }
}

// ======================= per-step kernels =======================

__global__ void k_layer1(int t, const u16t* __restrict__ Qb, const u16t* __restrict__ Kb,
                         const u16t* __restrict__ W1b, const float* __restrict__ b1f,
                         u16t* Hq, u16t* Hk, u16t* HTk, unsigned char* maskp) {
  __shared__ __attribute__((aligned(16))) u16t smem[2 * CH_ELEMS];
  int n0 = blockIdx.x * 64;
  int wv = threadIdx.x >> 5, lane = threadIdx.x & 31;
  int m0 = wv * 16;
  int r = m0 + (lane & 15);
  const u16t* arow = (r < BATCH)
      ? (Qb + (size_t)t * BATCH * DIM + (size_t)r * DIM)
      : (Kb + (size_t)t * BATCH * DIM + (size_t)(r - BATCH) * DIM);
  v8f acc[4];
#pragma unroll
  for (int j = 0; j < 4; ++j) zero8(acc[j]);
  mma_row64_lds<256>(arow, W1b, DIM, n0, threadIdx.x, DIM, smem, acc);
  int half = lane >> 4, nl = lane & 15;
#pragma unroll
  for (int j = 0; j < 4; ++j) {
    int n = n0 + j * 16 + nl;
    float bias = b1f[n];
#pragma unroll
    for (int e = 0; e < 8; ++e) {
      int m = m0 + e + 8 * half;
      float v = acc[j][e] + bias;
      if (m < BATCH) {
        Hq[(size_t)m * DIM + n] = f2bf(fmaxf(v, 0.0f));
      } else {
        int mk = m - BATCH;
        float rl = fmaxf(v, 0.0f);
        Hk[(size_t)mk * DIM + n] = f2bf(rl);
        HTk[(size_t)n * BATCH + mk] = f2bf(rl);
        maskp[(size_t)mk * DIM + n] = (v > 0.0f) ? 1 : 0;
      }
    }
  }
}

__global__ void k_layer2(int t, const u16t* __restrict__ Hq, const u16t* __restrict__ Hk,
                         const u16t* __restrict__ W2b, const float* __restrict__ b2f,
                         const float* __restrict__ Vf, float* out0, float* out1,
                         float* diff, float* lossPart) {
  __shared__ __attribute__((aligned(16))) u16t smem[2 * CH_ELEMS];
  __shared__ float red[256];
  int n0 = blockIdx.x * 64;
  int wv = threadIdx.x >> 5, lane = threadIdx.x & 31;
  int m0 = wv * 16;
  int r = m0 + (lane & 15);
  const u16t* arow = (r < BATCH) ? (Hq + (size_t)r * DIM)
                                 : (Hk + (size_t)(r - BATCH) * DIM);
  v8f acc[4];
#pragma unroll
  for (int j = 0; j < 4; ++j) zero8(acc[j]);
  mma_row64_lds<256>(arow, W2b, DIM, n0, threadIdx.x, DIM, smem, acc);
  float sq = 0.0f;
  int half = lane >> 4, nl = lane & 15;
#pragma unroll
  for (int j = 0; j < 4; ++j) {
    int n = n0 + j * 16 + nl;
    float bias = b2f[n];
#pragma unroll
    for (int e = 0; e < 8; ++e) {
      int m = m0 + e + 8 * half;
      float v = acc[j][e] + bias;
      if (m < BATCH) {
        size_t oi = (size_t)m * SEQ * DIM + (size_t)t * DIM + n;
        out0[oi] = v; out1[oi] = v;
      } else {
        int mk = m - BATCH;
        float d = v - Vf[(size_t)t * BATCH * DIM + (size_t)mk * DIM + n];
        diff[(size_t)mk * DIM + n] = d;
        sq += d * d;
      }
    }
  }
  red[threadIdx.x] = sq;
  __syncthreads();
#pragma unroll
  for (int s = 128; s; s >>= 1) {
    if (threadIdx.x < s) red[threadIdx.x] += red[threadIdx.x + s];
    __syncthreads();
  }
  if (threadIdx.x == 0) lossPart[blockIdx.x] = red[0];
}

__global__ void k_dout(const float* __restrict__ lossPart, const float* __restrict__ diff,
                       u16t* dout, u16t* doutT) {
  int i = blockIdx.x * 256 + threadIdx.x;
  float s = 0.0f;
#pragma unroll
  for (int p = 0; p < 8; ++p) s += lossPart[p];
  const float invN = 1.0f / (float)(BATCH * DIM);
  float loss = s * invN;
  float pt = (loss >= 1e-10f && loss <= 1e10f) ? 1.0f : 0.0f;
  float scale = pt * 2.0f * invN;
  if (i < BATCH * DIM) {
    float d = diff[i] * scale;
    int b = i >> 9, n = i & (DIM - 1);
    u16t h = f2bf(d);
    dout[i] = h;
    doutT[(size_t)n * BATCH + b] = h;
  }
}

__global__ void k_dpre(const u16t* __restrict__ dout, const u16t* __restrict__ W2Tb,
                       const unsigned char* __restrict__ maskp, u16t* dpre, u16t* dpreT) {
  __shared__ __attribute__((aligned(16))) u16t smem[2 * CH_ELEMS];
  int n0 = blockIdx.x * 64;
  int wv = threadIdx.x >> 5, lane = threadIdx.x & 31;
  int m0 = wv * 16;
  const u16t* arow = dout + (size_t)(m0 + (lane & 15)) * DIM;
  v8f acc[4];
#pragma unroll
  for (int j = 0; j < 4; ++j) zero8(acc[j]);
  mma_row64_lds<128>(arow, W2Tb, DIM, n0, threadIdx.x, DIM, smem, acc);
  int half = lane >> 4, nl = lane & 15;
#pragma unroll
  for (int j = 0; j < 4; ++j) {
    int n = n0 + j * 16 + nl;
#pragma unroll
    for (int e = 0; e < 8; ++e) {
      int m = m0 + e + 8 * half;
      float v = maskp[(size_t)m * DIM + n] ? acc[j][e] : 0.0f;
      u16t h = f2bf(v);
      dpre[(size_t)m * DIM + n] = h;
      dpreT[(size_t)n * BATCH + m] = h;
    }
  }
}

__global__ void k_bias(int t, const u16t* __restrict__ dout, const u16t* __restrict__ dpre,
                       const float* __restrict__ theta, const float* __restrict__ eta,
                       const float* __restrict__ alpha,
                       float* M1b, float* M2b, float* b1f, float* b2f) {
  int j = blockIdx.x * 256 + threadIdx.x;
  if (j >= DIM) return;
  float th = theta[t], et = eta[t], al = alpha[t];
  float g1 = 0.0f, g2 = 0.0f;
  for (int b = 0; b < BATCH; ++b) {
    g2 += bf2f(dout[(size_t)b * DIM + j]);
    g1 += bf2f(dpre[(size_t)b * DIM + j]);
  }
  float m1 = et * M1b[j] - th * g1; M1b[j] = m1; b1f[j] = (1.0f - al) * b1f[j] + m1;
  float m2 = et * M2b[j] - th * g2; M2b[j] = m2; b2f[j] = (1.0f - al) * b2f[j] + m2;
}

// gW (512x512, K=64) fused with momentum + decay; writes next step's bf16 copies.
// grid (8, 4, 2), 256 threads. z=0: W1 (A=dpre^T, B=k^T). z=1: W2 (A=dout^T, B=h^T).
__global__ void k_gw(int t, const u16t* __restrict__ dpreT, const u16t* __restrict__ doutT,
                     const u16t* __restrict__ KTb, const u16t* __restrict__ HTk,
                     const float* __restrict__ theta, const float* __restrict__ eta,
                     const float* __restrict__ alpha,
                     float* M1w, float* M2w, float* W1f, float* W2f,
                     u16t* W1b, u16t* W2b, u16t* W2Tb) {
  __shared__ __attribute__((aligned(16))) u16t smem[2 * CH_ELEMS];
  int sel = blockIdx.z;
  int n0 = blockIdx.x * 64;
  int wv = threadIdx.x >> 5, lane = threadIdx.x & 31;
  int m0 = blockIdx.y * 128 + wv * 16;
  const u16t* A  = sel ? doutT : dpreT;
  const u16t* Bm = sel ? HTk : (KTb + (size_t)t * BATCH * DIM);
  const u16t* arow = A + (size_t)(m0 + (lane & 15)) * BATCH;
  v8f acc[4];
#pragma unroll
  for (int j = 0; j < 4; ++j) zero8(acc[j]);
  mma_row64_lds<256>(arow, Bm, BATCH, n0, threadIdx.x, BATCH, smem, acc);  // K = 64
  float th = theta[t], et = eta[t], oma = 1.0f - alpha[t];
  float* Mw = sel ? M2w : M1w;
  float* Wf = sel ? W2f : W1f;
  u16t*  Wb = sel ? W2b : W1b;
  int half = lane >> 4, nl = lane & 15;
#pragma unroll
  for (int j = 0; j < 4; ++j) {
    int n = n0 + j * 16 + nl;
#pragma unroll
    for (int e = 0; e < 8; ++e) {
      int m = m0 + e + 8 * half;
      size_t idx = (size_t)m * DIM + n;
      float g = acc[j][e];
      float mv = et * Mw[idx] - th * g;  Mw[idx] = mv;
      float w  = oma * Wf[idx] + mv;     Wf[idx] = w;
      Wb[idx] = f2bf(w);
      if (sel) W2Tb[(size_t)n * DIM + m] = f2bf(w);
    }
  }
}

// ======================= host launcher =======================

extern "C" void kernel_launch(void* const* d_in, const int* in_sizes, int n_in,
                              void* d_out, int out_size, void* d_ws, size_t ws_size,
                              hipStream_t stream) {
  (void)in_sizes; (void)n_in; (void)out_size; (void)ws_size;
  const float* x  = (const float*)d_in[0];
  const float* WQ = (const float*)d_in[1];
  const float* WK = (const float*)d_in[2];
  const float* WV = (const float*)d_in[3];
  const float* tw = (const float*)d_in[4];  const float* tb = (const float*)d_in[5];
  const float* ew = (const float*)d_in[6];  const float* eb = (const float*)d_in[7];
  const float* aw = (const float*)d_in[8];  const float* ab = (const float*)d_in[9];
  const float* W1 = (const float*)d_in[10]; const float* b1 = (const float*)d_in[11];
  const float* W2 = (const float*)d_in[12]; const float* b2 = (const float*)d_in[13];
  float* out0 = (float*)d_out;
  float* out1 = out0 + (size_t)BATCH * SEQ * DIM;

  char* wsb = (char*)d_ws;
  size_t off = 0;
  auto alloc = [&](size_t bytes) -> void* {
    void* p = wsb + off;
    off = (off + bytes + 255) & ~(size_t)255;
    return p;
  };
  const size_t BTD = (size_t)BATCH * SEQ * DIM;
  const size_t DD  = (size_t)DIM * DIM;
  const size_t BD  = (size_t)BATCH * DIM;

  u16t* Xb   = (u16t*)alloc(BTD * 2);
  u16t* WQb  = (u16t*)alloc(DD * 2);
  u16t* WKb  = (u16t*)alloc(DD * 2);
  u16t* WVb  = (u16t*)alloc(DD * 2);
  u16t* Qb   = (u16t*)alloc(BTD * 2);
  u16t* Kb   = (u16t*)alloc(BTD * 2);
  u16t* KTb  = (u16t*)alloc(BTD * 2);
  float* Vf  = (float*)alloc(BTD * 4);
  float* theta = (float*)alloc(SEQ * 4);
  float* eta   = (float*)alloc(SEQ * 4);
  float* alpha = (float*)alloc(SEQ * 4);
  float* lossPart = (float*)alloc(8 * 4);
  float* W1f = (float*)alloc(DD * 4);
  float* W2f = (float*)alloc(DD * 4);
  float* M1w = (float*)alloc(DD * 4);
  float* M2w = (float*)alloc(DD * 4);
  u16t* W1b  = (u16t*)alloc(DD * 2);
  u16t* W2b  = (u16t*)alloc(DD * 2);
  u16t* W2Tb = (u16t*)alloc(DD * 2);
  float* b1f = (float*)alloc(DIM * 4);
  float* b2f = (float*)alloc(DIM * 4);
  float* M1b = (float*)alloc(DIM * 4);
  float* M2b = (float*)alloc(DIM * 4);
  u16t* Hq   = (u16t*)alloc(BD * 2);
  u16t* Hk   = (u16t*)alloc(BD * 2);
  u16t* HTk  = (u16t*)alloc(BD * 2);
  unsigned char* maskp = (unsigned char*)alloc(BD);
  float* diff = (float*)alloc(BD * 4);
  u16t* dout  = (u16t*)alloc(BD * 2);
  u16t* doutT = (u16t*)alloc(BD * 2);
  u16t* dpre  = (u16t*)alloc(BD * 2);
  u16t* dpreT = (u16t*)alloc(BD * 2);

  // ---- precompute (parallel over all of T) ----
  k_convert<<<(int)((BTD + 255) / 256), 256, 0, stream>>>(x, WQ, WK, WV, Xb, WQb, WKb, WVb);
  k_init<<<(int)((DD + 255) / 256), 256, 0, stream>>>(W1, W2, b1, b2, W1f, W2f, M1w, M2w,
                                                      W1b, W2b, W2Tb, b1f, b2f, M1b, M2b);
  k_gates<<<SEQ, 256, 0, stream>>>(x, tw, tb, ew, eb, aw, ab, theta, eta, alpha);
  k_qkv<<<dim3(DIM / 64, SEQ, 3), 128, 0, stream>>>(Xb, WQb, WKb, WVb, Qb, Kb, KTb, Vf);

  // ---- sequential scan over T (carry dependency enforced by stream order) ----
  for (int t = 0; t < SEQ; ++t) {
    k_layer1<<<DIM / 64, 256, 0, stream>>>(t, Qb, Kb, W1b, b1f, Hq, Hk, HTk, maskp);
    k_layer2<<<DIM / 64, 256, 0, stream>>>(t, Hq, Hk, W2b, b2f, Vf, out0, out1, diff, lossPart);
    k_dout<<<(int)((BD + 255) / 256), 256, 0, stream>>>(lossPart, diff, dout, doutT);
    k_dpre<<<DIM / 64, 128, 0, stream>>>(dout, W2Tb, maskp, dpre, dpreT);
    k_bias<<<(DIM + 255) / 256, 256, 0, stream>>>(t, dout, dpre, theta, eta, alpha,
                                                  M1b, M2b, b1f, b2f);
    k_gw<<<dim3(DIM / 64, 4, 2), 256, 0, stream>>>(t, dpreT, doutT, KTb, HTk,
                                                   theta, eta, alpha,
                                                   M1w, M2w, W1f, W2f, W1b, W2b, W2Tb);
  }
}
